// CausalSelfAttention_73332271611912
// MI455X (gfx1250) — matrix operand order repaired and checked by reading, not compiled, not run on hardware
//
#include <hip/hip_runtime.h>
#include <hip/hip_bf16.h>

// ---------------------------------------------------------------------------
// CDNA5 (gfx1250) causal self-attention, bf16 WMMA pipeline with async
// global->LDS staging (ASYNCcnt path) for the attention V tiles.
//   B=2, T=2048, C=1024, H=16, Dh=64
// ---------------------------------------------------------------------------

#define BATCH   2
#define T_SEQ   2048
#define C_MODEL 1024
#define NHEAD   16
#define HDIM    64
#define MROWS   (BATCH * T_SEQ)     // 4096
#define QKV_LD  (3 * C_MODEL)       // 3072

typedef __bf16 bf16_t;
typedef __attribute__((ext_vector_type(16))) __bf16          v16bf;
typedef __attribute__((ext_vector_type(16))) unsigned short  v16us;
typedef __attribute__((ext_vector_type(8)))  float           v8f;

// float -> bf16 (RNE via hardware convert)
static __device__ inline unsigned short f2bf(float f) {
    union { __bf16 b; unsigned short u; } cv;
    cv.b = (__bf16)f;
    return cv.u;
}

// 16-bit A-matrix 16x32 fragment, contiguous k order: VGPR pair p, lane-half -> k base = 16*half + 2*p.
static __device__ inline int a_kbase(int p, int half) {
    return half * 16 + 2 * p;
}

// Load an A fragment (16x32 bf16) from a row-major source.
// rowptr = start of this lane's row (row = lane%16), K0 = k offset.
static __device__ inline v16bf load_a_frag(const unsigned short* rowptr, int K0, int half) {
    v16us u;
#pragma unroll
    for (int p = 0; p < 8; ++p) {
        int kb = K0 + a_kbase(p, half);
        unsigned int w = *(const unsigned int*)(rowptr + kb);   // 2 contiguous bf16
        u[2 * p]     = (unsigned short)(w & 0xFFFFu);
        u[2 * p + 1] = (unsigned short)(w >> 16);
    }
    return __builtin_bit_cast(v16bf, u);
}

// B fragment (32x16 bf16): per lane 16 contiguous bf16 along K.
// p must point at element (n = lane%16, k = half*16), 4B-aligned at least.
static __device__ inline v16bf load_b_frag_contig(const unsigned short* p) {
    return __builtin_bit_cast(v16bf, *(const v16us*)p);
}

static __device__ inline v8f wmma_bf16(v16bf a, v16bf b, v8f c) {
    return __builtin_amdgcn_wmma_f32_16x16x32_bf16(false, a, false, b, (short)0, c, false, false);
}

// Async global -> LDS 16-byte copy (ASYNCcnt path, no VGPR data round-trip).
#define ASYNC_CP16(OFF)                                                        \
    asm volatile("global_load_async_to_lds_b128 %0, %1, off offset:" #OFF     \
                 :: "v"(lds_off), "v"(gsrc) : "memory")

#define WAIT_ASYNC0() asm volatile("s_wait_asynccnt 0x0" ::: "memory")

// ---------------------------------------------------------------------------
// Kernel 0: f32 -> bf16 conversion
// ---------------------------------------------------------------------------
__global__ void f32_to_bf16(const float* __restrict__ in, unsigned short* __restrict__ out, int n) {
    for (int i = blockIdx.x * blockDim.x + threadIdx.x; i < n; i += gridDim.x * blockDim.x)
        out[i] = f2bf(in[i]);
}

// ---------------------------------------------------------------------------
// Kernel 1/3: C[M,N] = A[M,K] * B[N,K]^T  (bf16 in, f32 acc)
// Wave: 16 rows x 64 cols (4 acc tiles). Block: 8 waves = 32 rows x 256 cols.
// Out either bf16 (Cb) or f32 (Cf).
// ---------------------------------------------------------------------------
__global__ __launch_bounds__(256) void gemm_bf16(
    const unsigned short* __restrict__ A,
    const unsigned short* __restrict__ Bw,
    unsigned short* __restrict__ Cb,
    float* __restrict__ Cf,
    int M, int N, int K) {
    const int lane = threadIdx.x & 31;
    const int wave = threadIdx.x >> 5;
    const int half = lane >> 4;
    const int ln   = lane & 15;
    const int m0 = blockIdx.y * 32 + (wave >> 2) * 16;
    const int n0 = blockIdx.x * 256 + (wave & 3) * 64;

    v8f acc[4] = {};
    const unsigned short* arow = A + (size_t)(m0 + ln) * K;

    for (int kk = 0; kk < K; kk += 32) {
        if (kk + 64 < K)
            __builtin_prefetch(arow + kk + 64, 0, 0);   // global_prefetch_b8
        v16bf af = load_a_frag(arow, kk, half);
#pragma unroll
        for (int nt = 0; nt < 4; ++nt) {
            const unsigned short* bp =
                Bw + (size_t)(n0 + nt * 16 + ln) * K + kk + half * 16;
            acc[nt] = wmma_bf16(af, load_b_frag_contig(bp), acc[nt]);
        }
    }

#pragma unroll
    for (int r = 0; r < 8; ++r) {
        const int row = m0 + r + 8 * half;
#pragma unroll
        for (int nt = 0; nt < 4; ++nt) {
            const int col = n0 + nt * 16 + ln;
            if (Cf) Cf[(size_t)row * N + col] = acc[nt][r];
            else    Cb[(size_t)row * N + col] = f2bf(acc[nt][r]);
        }
    }
}

// ---------------------------------------------------------------------------
// Kernel 2: causal flash attention over qkv (bf16), writes y (bf16, [B,T,C]).
// Grid: (T/128, B*H). Block: 256 = 8 waves; wave owns 16 query rows.
// V tiles staged to LDS with global_load_async_to_lds_b128 (ASYNCcnt),
// overlapping the S = Q*K^T WMMAs and the softmax.
// ---------------------------------------------------------------------------
#define V_STRIDE 72                       // halfwords per V row in LDS (144B, 16B aligned)
#define P_STRIDE 34                       // halfwords per P row in LDS
#define WAVE_LDS (32 * V_STRIDE + 16 * P_STRIDE)

__global__ __launch_bounds__(256) void attn_causal(
    const unsigned short* __restrict__ qkv,   // [B*T, 3072] bf16
    unsigned short* __restrict__ ybf) {       // [B*T, 1024] bf16
    __shared__ unsigned short lds[8 * WAVE_LDS];

    const int lane = threadIdx.x & 31;
    const int wave = threadIdx.x >> 5;
    const int half = lane >> 4;
    const int ln   = lane & 15;
    const int bh = blockIdx.y;
    const int b = bh >> 4, h = bh & 15;
    const int qbase = blockIdx.x * 128 + wave * 16;

    unsigned short* vtile = lds + wave * WAVE_LDS;
    unsigned short* ptile = vtile + 32 * V_STRIDE;

    // LDS byte address (addrspace(3) offset) for this lane's V-row slot:
    // generic LDS pointers carry the offset in the low 32 bits.
    const unsigned lds_off =
        (unsigned)(unsigned long long)(vtile + lane * V_STRIDE);

    // Q fragments (16 rows x 64) loaded once.
    const unsigned short* qrow =
        qkv + (size_t)(b * T_SEQ + qbase + ln) * QKV_LD + h * HDIM;
    const v16bf aq0 = load_a_frag(qrow, 0, half);
    const v16bf aq1 = load_a_frag(qrow, 32, half);

    float mrow[8], lrow[8], corr[8];
#pragma unroll
    for (int r = 0; r < 8; ++r) { mrow[r] = -1e30f; lrow[r] = 0.0f; }
    v8f acc[4] = {};

    const int nkb = ((qbase + 15) >> 5) + 1;   // 32-key blocks, causal bound
    for (int kb = 0; kb < nkb; ++kb) {
        const int key0 = kb * 32;

        // ---- async-stage V block (32 x 64 bf16): lane copies one row ------
        {
            const unsigned short* gsrc =
                qkv + (size_t)(b * T_SEQ + key0 + lane) * QKV_LD + 2 * C_MODEL + h * HDIM;
            ASYNC_CP16(0);  ASYNC_CP16(16); ASYNC_CP16(32); ASYNC_CP16(48);
            ASYNC_CP16(64); ASYNC_CP16(80); ASYNC_CP16(96); ASYNC_CP16(112);
        }

        // ---- S = Q * K^T for 32 keys: two 16x16 tiles, Dh=64 = 2 x K32 ----
        v8f s0 = {}, s1 = {};
#pragma unroll
        for (int c = 0; c < 2; ++c) {
            const int K0 = c * 32;
            const unsigned short* kp0 =
                qkv + (size_t)(b * T_SEQ + key0 + ln) * QKV_LD + C_MODEL + h * HDIM + K0 + half * 16;
            const unsigned short* kp1 = kp0 + (size_t)16 * QKV_LD;
            const v16bf aq = c ? aq1 : aq0;
            s0 = wmma_bf16(aq, load_b_frag_contig(kp0), s0);
            s1 = wmma_bf16(aq, load_b_frag_contig(kp1), s1);
        }

        // ---- scale, causal mask, online softmax (rows live in 16-lane halves) ----
        const float scale = 0.125f;   // 1/sqrt(64)
#pragma unroll
        for (int r = 0; r < 8; ++r) {
            float e0 = s0[r] * scale;
            float e1 = s1[r] * scale;
            const int q = qbase + r + 8 * half;
            if (key0 + ln      > q) e0 = -1e30f;
            if (key0 + 16 + ln > q) e1 = -1e30f;

            float mx = fmaxf(e0, e1);
#pragma unroll
            for (int off = 8; off; off >>= 1)
                mx = fmaxf(mx, __shfl_xor(mx, off, 32));
            const float mn = fmaxf(mrow[r], mx);
            const float cf = __expf(mrow[r] - mn);
            const float p0 = __expf(e0 - mn);
            const float p1 = __expf(e1 - mn);
            float ps = p0 + p1;
#pragma unroll
            for (int off = 8; off; off >>= 1)
                ps += __shfl_xor(ps, off, 32);
            lrow[r] = lrow[r] * cf + ps;
            mrow[r] = mn;
            corr[r] = cf;

            // park P (D-layout) into LDS so it can be re-read in A-layout
            const int prow = r + 8 * half;
            ptile[prow * P_STRIDE + ln]      = f2bf(p0);
            ptile[prow * P_STRIDE + 16 + ln] = f2bf(p1);
        }
#pragma unroll
        for (int nt = 0; nt < 4; ++nt)
#pragma unroll
            for (int r = 0; r < 8; ++r)
                acc[nt][r] *= corr[r];

        // ---- P as A fragment (16x32) from LDS ----
        v16bf pf;
        {
            v16us u;
#pragma unroll
            for (int p = 0; p < 8; ++p) {
                const int kb2 = a_kbase(p, half);
                unsigned int w = *(const unsigned int*)(ptile + ln * P_STRIDE + kb2);
                u[2 * p]     = (unsigned short)(w & 0xFFFFu);
                u[2 * p + 1] = (unsigned short)(w >> 16);
            }
            pf = __builtin_bit_cast(v16bf, u);
        }

        // ---- wait for the async V tile, then O += P * V -------------------
        WAIT_ASYNC0();
#pragma unroll
        for (int nt = 0; nt < 4; ++nt) {
            v16us u;
#pragma unroll
            for (int e = 0; e < 16; ++e)
                u[e] = vtile[(half * 16 + e) * V_STRIDE + nt * 16 + ln];
            acc[nt] = wmma_bf16(pf, __builtin_bit_cast(v16bf, u), acc[nt]);
        }
    }

    // ---- epilogue: normalize and write y[b, t, h*64 + d] as bf16 ----
#pragma unroll
    for (int r = 0; r < 8; ++r) {
        const float inv = 1.0f / lrow[r];
        const int row = qbase + r + 8 * half;
        const size_t base = (size_t)(b * T_SEQ + row) * C_MODEL + h * HDIM;
#pragma unroll
        for (int nt = 0; nt < 4; ++nt)
            ybf[base + nt * 16 + ln] = f2bf(acc[nt][r] * inv);
    }
}

// ---------------------------------------------------------------------------
// Host launcher
// ---------------------------------------------------------------------------
extern "C" void kernel_launch(void* const* d_in, const int* in_sizes, int n_in,
                              void* d_out, int out_size, void* d_ws, size_t ws_size,
                              hipStream_t stream) {
    const float* x      = (const float*)d_in[0];   // [2,2048,1024]
    const float* w_qkv  = (const float*)d_in[1];   // [3072,1024]
    const float* w_proj = (const float*)d_in[2];   // [1024,1024]
    float* out = (float*)d_out;                    // [2,2048,1024]

    unsigned short* ws      = (unsigned short*)d_ws;
    unsigned short* xbf     = ws;                                          // 4096*1024
    unsigned short* wqkvbf  = xbf    + (size_t)MROWS * C_MODEL;            // 3072*1024
    unsigned short* wprojbf = wqkvbf + (size_t)QKV_LD * C_MODEL;           // 1024*1024
    unsigned short* qkvbf   = wprojbf + (size_t)C_MODEL * C_MODEL;         // 4096*3072
    unsigned short* ybf     = qkvbf  + (size_t)MROWS * QKV_LD;             // 4096*1024

    f32_to_bf16<<<512, 256, 0, stream>>>(x,      xbf,     MROWS * C_MODEL);
    f32_to_bf16<<<512, 256, 0, stream>>>(w_qkv,  wqkvbf,  QKV_LD * C_MODEL);
    f32_to_bf16<<<256, 256, 0, stream>>>(w_proj, wprojbf, C_MODEL * C_MODEL);

    // qkv = x @ w_qkv^T   -> bf16
    dim3 g1(QKV_LD / 256, MROWS / 32);
    gemm_bf16<<<g1, 256, 0, stream>>>(xbf, wqkvbf, qkvbf, nullptr,
                                      MROWS, QKV_LD, C_MODEL);

    // flash attention -> y (bf16)
    attn_causal<<<dim3(T_SEQ / 128, BATCH * NHEAD), 256, 0, stream>>>(qkvbf, ybf);

    // out = y @ w_proj^T  -> f32
    dim3 g3(C_MODEL / 256, MROWS / 32);
    gemm_bf16<<<g3, 256, 0, stream>>>(ybf, wprojbf, nullptr, out,
                                      MROWS, C_MODEL, C_MODEL);
}
